// BiLSTM_CRF_27865747817287
// MI455X (gfx1250) — matrix operand order repaired and checked by reading, not compiled, not run on hardware
//
#include <hip/hip_runtime.h>
#include <hip/hip_bf16.h>

typedef __attribute__((ext_vector_type(16))) _Float16 v16h;
typedef __attribute__((ext_vector_type(8)))  float    v8f;

#define LQ   256   // sequence length
#define BQ   64    // batch
#define EQ   256   // embed dim
#define HHQ  256   // per-direction hidden
#define G4   1024  // 4*Hh
#define TQ   9     // tags

__device__ __forceinline__ float sigf(float x) {
    return 1.0f / (1.0f + __expf(-x));
}
__device__ __forceinline__ float tanhfast(float x) {
    float xc = fminf(fmaxf(x, -15.0f), 15.0f);
    float e = __expf(2.0f * xc);
    return (e - 1.0f) / (e + 1.0f);
}

// ---------------------------------------------------------------------------
// Pack the four [1024,256] weight matrices into WMMA B-fragment order (f16):
// PW[mat][kt][nt][lane][h], element = W[n][k], n = nt*16+(lane&15),
// k = kt*32 + (h<8 ? (lane>>4)*8+h : 16+(lane>>4)*8+(h-8)).  mat: 0=ih_f 1=ih_b 2=hh_f 3=hh_b
// ---------------------------------------------------------------------------
__global__ void pack_w_kernel(const float* __restrict__ wihf,
                              const float* __restrict__ wihb,
                              const float* __restrict__ whhf,
                              const float* __restrict__ whhb,
                              _Float16* __restrict__ pw) {
    int idx  = blockIdx.x * 256 + threadIdx.x;           // 2^20 threads
    int h    = idx & 15;
    int lane = (idx >> 4) & 31;
    int nt   = (idx >> 9) & 63;
    int kt   = (idx >> 15) & 7;
    int mat  = idx >> 18;
    int n = nt * 16 + (lane & 15);
    int k = kt * 32 + ((h < 8) ? (((lane >> 4) << 3) + h) : (16 + ((lane >> 4) << 3) + (h - 8)));
    const float* src = (mat == 0) ? wihf : (mat == 1) ? wihb
                     : (mat == 2) ? whhf : whhb;
    pw[idx] = (_Float16)src[(size_t)n * EQ + k];
}

// bias_sum[dir][n] = b_ih[n] + b_hh[n]
__global__ void pack_bias_kernel(const float* __restrict__ bif,
                                 const float* __restrict__ bhf,
                                 const float* __restrict__ bib,
                                 const float* __restrict__ bhb,
                                 float* __restrict__ bias) {
    int idx = blockIdx.x * 256 + threadIdx.x;
    if (idx >= 2 * G4) return;
    int dir = idx >> 10, n = idx & 1023;
    bias[idx] = dir ? (bib[n] + bhb[n]) : (bif[n] + bhf[n]);
}

// ---------------------------------------------------------------------------
// Embedding gather -> X packed in WMMA A-fragment order (f16):
// PX[mt][kt][lane][h]; row m = mt*16+(lane&15) = l*64+b;
// K = kt*32 + (h<8 ? (lane>>4)*8+h : 16+(lane>>4)*8+(h-8))
// ---------------------------------------------------------------------------
__global__ void pack_x_kernel(const int* __restrict__ sent,
                              const float* __restrict__ emb,
                              _Float16* __restrict__ px) {
    int idx  = blockIdx.x * 256 + threadIdx.x;           // 2^22 threads
    int h    = idx & 15;
    int lane = (idx >> 4) & 31;
    int kt   = (idx >> 9) & 7;
    int mt   = idx >> 12;
    int m = mt * 16 + (lane & 15);
    int l = m >> 6, b = m & 63;
    int kbase = (lane >> 4) * 8;
    int kp = (h < 8) ? (kbase + h) : (16 + kbase + (h - 8));
    int e  = kt * 32 + kp;
    int tok = sent[b * LQ + l];
    px[idx] = (_Float16)emb[(size_t)tok * EQ + e];
}

// ---------------------------------------------------------------------------
// Pre-gate GEMM: PREG[dir] = X(16384x256) @ W_ih^T(256x1024) + (b_ih+b_hh)
// f16 WMMA, f32 accum.  Each wave: 16 rows x 64 cols (4 tiles, K-loop of 8).
// Output stored in D-fragment packed order: [Mtile][Ntile][lane][8 f32].
// ---------------------------------------------------------------------------
__global__ __launch_bounds__(256) void pregate_gemm_kernel(
        const _Float16* __restrict__ px,
        const _Float16* __restrict__ pwih,   // 2 dirs contiguous
        const float*    __restrict__ biassum,
        float*          __restrict__ preg) {
    int wid  = blockIdx.x * 8 + (threadIdx.x >> 5);
    int lane = threadIdx.x & 31;
    int dir  = wid >> 14;
    int rem  = wid & 16383;
    int mt   = rem >> 4;      // 0..1023
    int ng   = rem & 15;      // group of 4 N-tiles

    const v16h* A  = (const v16h*)px;
    const v16h* Bm = (const v16h*)(pwih + (size_t)dir * (8 * 64 * 32 * 16));

    v8f acc[4];
#pragma unroll
    for (int j = 0; j < 4; ++j) {
        int n = (ng * 4 + j) * 16 + (lane & 15);
        float bv = biassum[dir * G4 + n];
#pragma unroll
        for (int v = 0; v < 8; ++v) acc[j][v] = bv;
    }

    for (int kt = 0; kt < 8; ++kt) {
        v16h a = A[((size_t)mt * 8 + kt) * 32 + lane];
        if (kt < 7)
            __builtin_prefetch(&A[((size_t)mt * 8 + kt + 1) * 32 + lane], 0, 1);
#pragma unroll
        for (int j = 0; j < 4; ++j) {
            int nt = ng * 4 + j;
            v16h b = Bm[((size_t)kt * 64 + nt) * 32 + lane];
            acc[j] = __builtin_amdgcn_wmma_f32_16x16x32_f16(
                false, a, false, b, (short)0, acc[j], false, false);
        }
    }

    v8f* out = (v8f*)preg + (size_t)dir * (1024ull * 64 * 32);
#pragma unroll
    for (int j = 0; j < 4; ++j) {
        int nt = ng * 4 + j;
        out[((size_t)mt * 64 + nt) * 32 + lane] = acc[j];
    }
}

// ---------------------------------------------------------------------------
// Recurrent BiLSTM. grid=2 (blockIdx.x = direction), 1024 threads = 32 waves.
// LDS: h double-buffer (2 x 32KB, A-fragment packed f16) +
//      W_hh K-slice triple-buffer (3 x 64KB) streamed via
//      GLOBAL_LOAD_ASYNC_TO_LDS_B128 (ASYNCcnt).
// Each wave owns 2 (batch-tile, hh-tile) jobs = 8 accumulators (i/f/g/o x 2).
// Gate nonlinearity + c update run fully in registers (D-frag layout keeps
// i,f,g,o of one (b,hh) element in the same lane).
// ---------------------------------------------------------------------------
__global__ __launch_bounds__(1024) void lstm_kernel(
        const float*    __restrict__ preg,
        const _Float16* __restrict__ pwhh,   // 2 dirs contiguous
        const float*    __restrict__ h0,
        const float*    __restrict__ c0,
        float*          __restrict__ hstore) {
    extern __shared__ char smem[];
    // [0, 32K)       hbuf0
    // [32K, 64K)     hbuf1
    // [64K, 64K+192K) wbuf slots 0..2 (64KB each)

    const int dir  = blockIdx.x;
    const int tid  = threadIdx.x;
    const int lane = tid & 31;
    const int wave = tid >> 5;

    const int j0  = wave * 2;
    const int bt  = j0 >> 4;           // batch tile, same for both jobs
    const int ht0 = j0 & 15;           // hh tile of job 0
    const int ht1 = (j0 + 1) & 15;     // hh tile of job 1

    const _Float16* pwdir = pwhh + (size_t)dir * (8 * 64 * 32 * 16);
    const v8f* pregv = (const v8f*)preg + (size_t)dir * (1024ull * 64 * 32);

    const uint32_t ldsbase = (uint32_t)(uintptr_t)smem;   // LDS byte offset
    const uint32_t wbufOff = ldsbase + 65536;

    // ---- init: c into registers, h0 scattered into hbuf0 (A-frag layout)
    float creg[2][8];
    {
        int r = (lane >> 4) << 3;
#pragma unroll
        for (int j = 0; j < 2; ++j) {
            int htj = j ? ht1 : ht0;
            int hh  = htj * 16 + (lane & 15);
#pragma unroll
            for (int v = 0; v < 8; ++v) {
                int b = bt * 16 + r + v;
                creg[j][v] = c0[(dir * BQ + b) * HHQ + hh];
                float hv   = h0[(dir * BQ + b) * HHQ + hh];
                int rr = b & 15;
                int kt2 = hh >> 5, kp = hh & 31, grp = kp >> 3;
                int ln = rr + ((grp & 1) << 4);
                int hi = (kp & 7) + ((grp >> 1) << 3);
                ((_Float16*)smem)[(((b >> 4) * 8 + kt2) * 32 + ln) * 16 + hi] =
                    (_Float16)hv;
            }
        }
    }
    __syncthreads();

    for (int t = 0; t < LQ; ++t) {
        int row = dir ? (LQ - 1 - t) : t;
        const v16h* hb    = (const v16h*)(smem + (t & 1) * 32768);
        _Float16*   hnext = (_Float16*)(smem + ((t & 1) ^ 1) * 32768);

        // accumulators init from pre-gates (D-fragment packed)
        v8f acc[2][4];
#pragma unroll
        for (int j = 0; j < 2; ++j) {
            int htj = j ? ht1 : ht0;
#pragma unroll
            for (int g = 0; g < 4; ++g)
                acc[j][g] =
                    pregv[((size_t)(row * 4 + bt) * 64 + (g * 16 + htj)) * 32 + lane];
        }

        // prologue: async-stage W slice kt=0 into slot 0 (4 x b128 per thread)
#pragma unroll
        for (int c2 = 0; c2 < 4; ++c2) {
            uint32_t ldsa = wbufOff + c2 * 16384 + tid * 16;
            uint32_t goff = c2 * 16384 + tid * 16;
            asm volatile("global_load_async_to_lds_b128 %0, %1, %2"
                         :: "v"(ldsa), "v"(goff), "s"(pwdir) : "memory");
        }

        for (int kt = 0; kt < 8; ++kt) {
            if (kt < 7) {
                int slot = (kt + 1) % 3;
#pragma unroll
                for (int c2 = 0; c2 < 4; ++c2) {
                    uint32_t ldsa = wbufOff + slot * 65536 + c2 * 16384 + tid * 16;
                    uint32_t goff = (kt + 1) * 65536 + c2 * 16384 + tid * 16;
                    asm volatile("global_load_async_to_lds_b128 %0, %1, %2"
                                 :: "v"(ldsa), "v"(goff), "s"(pwdir) : "memory");
                }
                asm volatile("s_wait_asynccnt 0x4" ::: "memory");
            } else {
                asm volatile("s_wait_asynccnt 0x0" ::: "memory");
            }
            __syncthreads();

            const v16h* wb = (const v16h*)(smem + 65536 + (kt % 3) * 65536);
            v16h a = hb[(bt * 8 + kt) * 32 + lane];
#pragma unroll
            for (int j = 0; j < 2; ++j) {
                int htj = j ? ht1 : ht0;
#pragma unroll
                for (int g = 0; g < 4; ++g) {
                    v16h bfr = wb[(g * 16 + htj) * 32 + lane];
                    acc[j][g] = __builtin_amdgcn_wmma_f32_16x16x32_f16(
                        false, a, false, bfr, (short)0, acc[j][g], false, false);
                }
            }
        }

        // in-register gate nonlinearity + c/h update
        {
            int r = (lane >> 4) << 3;
#pragma unroll
            for (int j = 0; j < 2; ++j) {
                int htj = j ? ht1 : ht0;
                int hh  = htj * 16 + (lane & 15);
#pragma unroll
                for (int v = 0; v < 8; ++v) {
                    int b = bt * 16 + r + v;
                    float ig = acc[j][0][v], fg = acc[j][1][v];
                    float gg = acc[j][2][v], og = acc[j][3][v];
                    float c = sigf(fg) * creg[j][v] + sigf(ig) * tanhfast(gg);
                    creg[j][v] = c;
                    float h = sigf(og) * tanhfast(c);
                    hstore[((size_t)dir * LQ + row) * BQ * HHQ + b * HHQ + hh] = h;
                    int rr = b & 15;
                    int kt2 = hh >> 5, kp = hh & 31, grp = kp >> 3;
                    int ln = rr + ((grp & 1) << 4);
                    int hi = (kp & 7) + ((grp >> 1) << 3);
                    hnext[(((b >> 4) * 8 + kt2) * 32 + ln) * 16 + hi] = (_Float16)h;
                }
            }
        }
        __syncthreads();
    }
}

// ---------------------------------------------------------------------------
// feats[l][b][t] = [hf|hb] . w_out[t] + b_out[t]
// ---------------------------------------------------------------------------
__global__ void feats_kernel(const float* __restrict__ hst,
                             const float* __restrict__ w_out,
                             const float* __restrict__ b_out,
                             float* __restrict__ feats) {
    int idx = blockIdx.x * 256 + threadIdx.x;
    if (idx >= LQ * BQ * TQ) return;
    int tt = idx % TQ;
    int lb = idx / TQ;                       // l*64 + b
    const float* hf = hst + (size_t)lb * HHQ;
    const float* hb = hst + (size_t)LQ * BQ * HHQ + (size_t)lb * HHQ;
    const float* wf = w_out + tt * 512;
    float s = b_out[tt];
    for (int h = 0; h < HHQ; ++h) s += hf[h] * wf[h];
    for (int h = 0; h < HHQ; ++h) s += hb[h] * wf[HHQ + h];
    feats[idx] = s;
}

// ---------------------------------------------------------------------------
// CRF Viterbi: one thread per batch element, score[9] in registers,
// backpointers to global, then backtrace.  Output tags as float [B][L].
// ---------------------------------------------------------------------------
__global__ void viterbi_kernel(const float* __restrict__ feats,
                               const unsigned char* __restrict__ mask,
                               const float* __restrict__ st,
                               const float* __restrict__ et,
                               const float* __restrict__ trans,
                               int* __restrict__ hist,
                               float* __restrict__ out) {
    __shared__ float tr[TQ * TQ];
    int tid = threadIdx.x;
    for (int i = tid; i < TQ * TQ; i += 64) tr[i] = trans[i];
    __syncthreads();
    int b = tid;
    float sc[TQ];
#pragma unroll
    for (int j = 0; j < TQ; ++j) sc[j] = st[j] + feats[b * TQ + j];

    for (int t = 1; t < LQ; ++t) {
        bool m = mask[b * LQ + t] != 0;
        float ns[TQ];
        int   bp[TQ];
#pragma unroll
        for (int cur = 0; cur < TQ; ++cur) {
            float best = sc[0] + tr[0 * TQ + cur];
            int   arg  = 0;
#pragma unroll
            for (int p = 1; p < TQ; ++p) {
                float v = sc[p] + tr[p * TQ + cur];
                if (v > best) { best = v; arg = p; }
            }
            ns[cur] = best + feats[(t * BQ + b) * TQ + cur];
            bp[cur] = arg;
        }
#pragma unroll
        for (int cur = 0; cur < TQ; ++cur) {
            hist[((t - 1) * BQ + b) * TQ + cur] = bp[cur];
            sc[cur] = m ? ns[cur] : sc[cur];
        }
    }

    float bestf = sc[0] + et[0];
    int tag = 0;
#pragma unroll
    for (int j = 1; j < TQ; ++j) {
        float v = sc[j] + et[j];
        if (v > bestf) { bestf = v; tag = j; }
    }
    out[b * LQ + (LQ - 1)] = (float)tag;
    for (int t = LQ - 2; t >= 0; --t) {
        int pv = hist[(t * BQ + b) * TQ + tag];
        if (mask[b * LQ + (t + 1)] != 0) tag = pv;
        out[b * LQ + t] = (float)tag;
    }
}

// ---------------------------------------------------------------------------
extern "C" void kernel_launch(void* const* d_in, const int* in_sizes, int n_in,
                              void* d_out, int out_size, void* d_ws, size_t ws_size,
                              hipStream_t stream) {
    const int*   sentence = (const int*)d_in[0];
    const unsigned char* mask = (const unsigned char*)d_in[1];
    const float* embed  = (const float*)d_in[2];
    const float* w_ih_f = (const float*)d_in[3];
    const float* w_hh_f = (const float*)d_in[4];
    const float* b_ih_f = (const float*)d_in[5];
    const float* b_hh_f = (const float*)d_in[6];
    const float* w_ih_b = (const float*)d_in[7];
    const float* w_hh_b = (const float*)d_in[8];
    const float* b_ih_b = (const float*)d_in[9];
    const float* b_hh_b = (const float*)d_in[10];
    const float* h0     = (const float*)d_in[11];
    const float* c0     = (const float*)d_in[12];
    const float* w_out  = (const float*)d_in[13];
    const float* b_out  = (const float*)d_in[14];
    const float* s_tr   = (const float*)d_in[15];
    const float* e_tr   = (const float*)d_in[16];
    const float* trans  = (const float*)d_in[17];

    // workspace layout
    char* ws = (char*)d_ws;
    size_t off = 0;
    _Float16* PX   = (_Float16*)(ws + off); off += 8388608;    // X packed f16
    _Float16* PW   = (_Float16*)(ws + off); off += 2097152;    // 4 matrices packed f16
    float*    BIAS = (float*)   (ws + off); off += 8192;       // [2][1024]
    float*    PREG = (float*)   (ws + off); off += 134217728;  // pre-gates, D-packed
    float*    HST  = (float*)   (ws + off); off += 33554432;   // h states [2][L][B][Hh]
    float*    FEAT = (float*)   (ws + off); off += 589824;     // [L][B][T]
    int*      HIST = (int*)     (ws + off); off += 587520;     // [L-1][B][T]

    pack_w_kernel<<<4096, 256, 0, stream>>>(w_ih_f, w_ih_b, w_hh_f, w_hh_b, PW);
    pack_bias_kernel<<<8, 256, 0, stream>>>(b_ih_f, b_hh_f, b_ih_b, b_hh_b, BIAS);
    pack_x_kernel<<<16384, 256, 0, stream>>>(sentence, embed, PX);

    pregate_gemm_kernel<<<4096, 256, 0, stream>>>(PX, PW, BIAS, PREG);

    const size_t lstm_smem = 65536 + 3 * 65536;  // 256 KB
    hipFuncSetAttribute((const void*)lstm_kernel,
                        hipFuncAttributeMaxDynamicSharedMemorySize,
                        (int)lstm_smem);
    lstm_kernel<<<2, 1024, lstm_smem, stream>>>(
        PREG, PW + 2 * 262144, h0, c0, HST);

    feats_kernel<<<576, 256, 0, stream>>>(HST, w_out, b_out, FEAT);

    viterbi_kernel<<<1, 64, 0, stream>>>(FEAT, mask, s_tr, e_tr, trans,
                                         HIST, (float*)d_out);
}
